// Tacotron2Decoder_26414048871082
// MI455X (gfx1250) — matrix operand order repaired and checked by reading, not compiled, not run on hardware
//
#include <hip/hip_runtime.h>
#include <hip/hip_bf16.h>
#include <math.h>

// ---------------------------------------------------------------------------
// Tacotron2 decoder for MI455X (gfx1250, wave32, WMMA).
// Strategy: f16 weights (fit in 192MB L2), fp32 accumulate via
// v_wmma_f32_16x16x32_f16. M=32 batch = 2 WMMA M-tiles. LSTM gate fusion
// in-block. Sequential 500-step loop of stream-ordered kernels.
// ---------------------------------------------------------------------------

typedef __attribute__((ext_vector_type(16))) _Float16 v16h;
typedef __attribute__((ext_vector_type(8)))  _Float16 v8h;
typedef __attribute__((ext_vector_type(8)))  float    v8f;

#define BATCH 32
#define T_OUT 500
#define T_IN  200
#define N_MEL 80
#define ENC   512
#define PRE_D 256
#define ARNN  1024
#define DRNN  1024
#define ATTN_DIM 128
#define LOC_F 32
#define LOC_K 31
#define LDS_STRIDE 512   // chunked K staging width (halfs)

__device__ __forceinline__ float sigmoidf_(float x) { return 1.0f / (1.0f + expf(-x)); }

// Build A fragment (16x32 f16, MxK) from LDS staging buffer.
// Layout used here (same k order as B packing): lane L holds row M=L&15; K = slot (contiguous 0..15),
// lanes>=16 carry K offset +16.  => two contiguous 8-half (16B) LDS reads per lane.
__device__ __forceinline__ v16h load_a_frag(const _Float16* lds, int mbase, int kk, int lane) {
    int m  = mbase + (lane & 15);
    int kb = (kk << 5) + ((lane & 16) ? 16 : 0);
    const v8h lo = *(const v8h*)(lds + m * LDS_STRIDE + kb);
    const v8h hi = *(const v8h*)(lds + m * LDS_STRIDE + kb + 8);
    v16h a;
#pragma unroll
    for (int i = 0; i < 8; i++) { a[i] = lo[i]; a[8 + i] = hi[i]; }
    return a;
}

// ---------------------------------------------------------------------------
// Weight conversion fp32 -> f16, fragment-major tiling.
// Storage: frag(kt,nt) at ((nt*KT+kt)*512), element (lane,slot) at lane*16+slot.
// B layout: N = nt*16 + (lane&15); K = kt*32 + ((lane&16)?16:0) + slot.
// mode 0: row-concat [src0(rows0) ; src1(rows1)] both width-padded to tile N.
// mode 1: col-concat [src0(ncols0) | src1(ncols1) | zero-pad].
// ---------------------------------------------------------------------------
__global__ __launch_bounds__(256) void cvt_weights(
    _Float16* __restrict__ dst,
    const float* __restrict__ src0, int rows0, int ncols0,
    const float* __restrict__ src1, int rows1, int ncols1,
    int KT, int mode, long total)
{
    for (long e = (long)blockIdx.x * 256 + threadIdx.x; e < total;
         e += (long)gridDim.x * 256) {
        int s    = (int)(e & 15);
        int lane = (int)((e >> 4) & 31);
        long frag = e >> 9;
        int kt = (int)(frag % KT);
        int nt = (int)(frag / KT);
        int k = kt * 32 + ((lane & 16) ? 16 : 0) + s;
        int n = nt * 16 + (lane & 15);
        float v = 0.0f;
        if (mode == 0) {
            if (k < rows0) { if (n < ncols0) v = src0[(long)k * ncols0 + n]; }
            else if (k < rows0 + rows1) { if (n < ncols1) v = src1[(long)(k - rows0) * ncols1 + n]; }
        } else {
            if (k < rows0) {
                if (n < ncols0) v = src0[(long)k * ncols0 + n];
                else if (src1 && n < ncols0 + ncols1) v = src1[(long)k * ncols1 + (n - ncols0)];
            }
        }
        dst[e] = (_Float16)v;
    }
}

__global__ __launch_bounds__(256) void zero_f32(float* __restrict__ p, long n) {
    for (long i = (long)blockIdx.x * 256 + threadIdx.x; i < n; i += (long)gridDim.x * 256)
        p[i] = 0.0f;
}

// ---------------------------------------------------------------------------
// Generic WMMA GEMM: C[M,N] = (relu?) A[M,Ksrc (zero-pad to Ktot)] @ W(tiled f16).
// grid.x = M/32 row-blocks, grid.y*8 waves = N/16 col-tiles. block = 256 (8 waves).
// a_mode 0: plain rows.  a_mode 1: shifted-mel rows (teacher forcing go-frame).
// ---------------------------------------------------------------------------
__global__ __launch_bounds__(256) void gemm_wmma(
    const _Float16* __restrict__ W, int KT, int Ktot,
    const float* __restrict__ A, int ldA, int Ksrc, int a_mode,
    float* __restrict__ C, int ldC, int relu)
{
    __shared__ __align__(16) _Float16 ldsA[32 * LDS_STRIDE];
    const int tid = threadIdx.x, lane = tid & 31, w = tid >> 5;
    const int m0 = blockIdx.x * 32;
    const int nt = blockIdx.y * 8 + w;
    v8f acc0 = {}, acc1 = {};

    for (int k0 = 0; k0 < Ktot; k0 += LDS_STRIDE) {
        const int ch = min(LDS_STRIDE, Ktot - k0);
        for (int idx = tid; idx < 32 * ch; idx += 256) {
            int m = idx / ch, k = idx - m * ch, g = k0 + k;
            int r = m0 + m;
            float v = 0.0f;
            if (a_mode == 0) {
                if (g < Ksrc) v = A[(long)r * ldA + g];
            } else {
                int b = r / T_OUT, tt = r - b * T_OUT;
                if (tt > 0 && g < Ksrc) v = A[(long)(b * T_OUT + tt - 1) * ldA + g];
            }
            ldsA[m * LDS_STRIDE + k] = (_Float16)v;
        }
        __syncthreads();
        const int kiters = ch >> 5;
        for (int kk = 0; kk < kiters; kk++) {
            const int kt = (k0 >> 5) + kk;
            v16h a0 = load_a_frag(ldsA, 0,  kk, lane);
            v16h a1 = load_a_frag(ldsA, 16, kk, lane);
            v16h bf = *(const v16h*)(W + (((long)nt * KT + kt) << 9) + lane * 16);
            acc0 = __builtin_amdgcn_wmma_f32_16x16x32_f16(false, a0, false, bf, (short)0, acc0, false, false);
            acc1 = __builtin_amdgcn_wmma_f32_16x16x32_f16(false, a1, false, bf, (short)0, acc1, false, false);
        }
        __syncthreads();
    }
    const int n  = nt * 16 + (lane & 15);
    const int mb = (lane & 16) ? 8 : 0;
#pragma unroll
    for (int r = 0; r < 8; r++) {
        float v0 = acc0[r], v1 = acc1[r];
        if (relu) { v0 = fmaxf(v0, 0.0f); v1 = fmaxf(v1, 0.0f); }
        C[(long)(m0 + mb + r)      * ldC + n] = v0;
        C[(long)(m0 + 16 + mb + r) * ldC + n] = v1;
    }
}

// ---------------------------------------------------------------------------
// Fused LSTM step: z = [s0;s1;s2] @ Wcomb + bias -> gates -> h,c update.
// N=4096 gates. 32 WGs; WG wg owns hidden units [wg*32, wg*32+32) across all
// 4 gates (8 waves -> (gate, half16) col-tiles), so the nonlinearity fuses
// in-block via a small z LDS buffer.
// ---------------------------------------------------------------------------
__global__ __launch_bounds__(256) void lstm_step(
    const _Float16* __restrict__ W, int KT, int Ktot,
    const float* __restrict__ s0, int ld0, int L0,
    const float* __restrict__ s1, int ld1, int L1,
    const float* __restrict__ s2, int ld2, int L2,
    const float* __restrict__ bias,
    float* __restrict__ cstate, float* __restrict__ hout)
{
    __shared__ __align__(16) _Float16 ldsA[32 * LDS_STRIDE];
    __shared__ float zbuf[32 * 128];
    const int tid = threadIdx.x, lane = tid & 31, w = tid >> 5;
    const int wg = blockIdx.x;
    const int gidx = w >> 1, hh = w & 1;
    const int nt = gidx * 64 + wg * 2 + hh;   // col tile: gate*1024 + wg*32 + hh*16
    v8f acc0 = {}, acc1 = {};

    for (int k0 = 0; k0 < Ktot; k0 += LDS_STRIDE) {
        const int ch = min(LDS_STRIDE, Ktot - k0);
        for (int idx = tid; idx < 32 * ch; idx += 256) {
            int m = idx / ch, k = idx - m * ch, g = k0 + k;
            float v;
            if (g < L0)            v = s0[(long)m * ld0 + g];
            else if (g < L0 + L1)  v = s1[(long)m * ld1 + (g - L0)];
            else                   v = s2[(long)m * ld2 + (g - L0 - L1)];
            ldsA[m * LDS_STRIDE + k] = (_Float16)v;
        }
        __syncthreads();
        const int kiters = ch >> 5;
        for (int kk = 0; kk < kiters; kk++) {
            const int kt = (k0 >> 5) + kk;
            v16h a0 = load_a_frag(ldsA, 0,  kk, lane);
            v16h a1 = load_a_frag(ldsA, 16, kk, lane);
            v16h bf = *(const v16h*)(W + (((long)nt * KT + kt) << 9) + lane * 16);
            acc0 = __builtin_amdgcn_wmma_f32_16x16x32_f16(false, a0, false, bf, (short)0, acc0, false, false);
            acc1 = __builtin_amdgcn_wmma_f32_16x16x32_f16(false, a1, false, bf, (short)0, acc1, false, false);
        }
        __syncthreads();
    }
    // z -> LDS (local col = w*16 + n == gate*32 + unit)
    const int lc = w * 16 + (lane & 15);
    const int mb = (lane & 16) ? 8 : 0;
#pragma unroll
    for (int r = 0; r < 8; r++) {
        zbuf[(mb + r)      * 128 + lc] = acc0[r];
        zbuf[(mb + 16 + r) * 128 + lc] = acc1[r];
    }
    __syncthreads();
    // gate fusion: 32 rows x 32 units
    for (int i = tid; i < 1024; i += 256) {
        int m = i >> 5, u = i & 31;
        int j = wg * 32 + u;
        float zi = zbuf[m * 128 +      u] + bias[j];
        float zf = zbuf[m * 128 + 32 + u] + bias[1024 + j];
        float zg = zbuf[m * 128 + 64 + u] + bias[2048 + j];
        float zo = zbuf[m * 128 + 96 + u] + bias[3072 + j];
        float c_old = cstate[m * 1024 + j];
        float c2 = sigmoidf_(zf) * c_old + sigmoidf_(zi) * tanhf(zg);
        float h2 = sigmoidf_(zo) * tanhf(c2);
        cstate[m * 1024 + j] = c2;
        hout[m * 1024 + j]   = h2;
    }
}

// ---------------------------------------------------------------------------
// Location-sensitive attention: one WG per batch row. Small VALU work.
// ---------------------------------------------------------------------------
__global__ __launch_bounds__(256) void attn_step(
    const float* __restrict__ h,       // new h_a [32,1024]
    float* __restrict__ aw, float* __restrict__ acum,
    const float* __restrict__ pm,      // processed_memory [32,200,128]
    const float* __restrict__ memory,  // [32,200,512]
    const float* __restrict__ Wq,      // [1024,128]
    const float* __restrict__ convw,   // [31,2,32]
    const float* __restrict__ Wloc,    // [32,128]
    const float* __restrict__ vvec,    // [128]
    float* __restrict__ ctx,           // [32,512]
    float* __restrict__ aligns,        // base of aligns output
    int t)
{
    __shared__ float sh_h[1024], sh_aw[T_IN], sh_ac[T_IN], sh_q[ATTN_DIM];
    __shared__ float sh_f[T_IN * LOC_F], sh_e[T_IN], sh_awn[T_IN];
    __shared__ float s_max, s_sum;
    const int b = blockIdx.x, tid = threadIdx.x;

    for (int k = tid; k < 1024; k += 256) sh_h[k] = h[b * 1024 + k];
    for (int k = tid; k < T_IN; k += 256) { sh_aw[k] = aw[b * T_IN + k]; sh_ac[k] = acum[b * T_IN + k]; }
    __syncthreads();

    if (tid < ATTN_DIM) {
        float s = 0.0f;
        for (int k = 0; k < 1024; k++) s += sh_h[k] * Wq[k * ATTN_DIM + tid];
        sh_q[tid] = s;
    }
    for (int idx = tid; idx < T_IN * LOC_F; idx += 256) {
        int tt = idx / LOC_F, c = idx - tt * LOC_F;
        float s = 0.0f;
        for (int wdx = 0; wdx < LOC_K; wdx++) {
            int src = tt + wdx - (LOC_K / 2);
            if (src >= 0 && src < T_IN)
                s += sh_aw[src] * convw[(wdx * 2 + 0) * LOC_F + c]
                   + sh_ac[src] * convw[(wdx * 2 + 1) * LOC_F + c];
        }
        sh_f[tt * LOC_F + c] = s;
    }
    __syncthreads();

    if (tid < T_IN) {
        const float* pmr = pm + ((long)(b * T_IN + tid)) * ATTN_DIM;
        float e = 0.0f;
        for (int d = 0; d < ATTN_DIM; d++) {
            float s = sh_q[d] + pmr[d];
#pragma unroll
            for (int c = 0; c < LOC_F; c++) s += sh_f[tid * LOC_F + c] * Wloc[c * ATTN_DIM + d];
            e += tanhf(s) * vvec[d];
        }
        sh_e[tid] = e;
    }
    __syncthreads();
    if (tid == 0) {
        float mx = sh_e[0];
        for (int k = 1; k < T_IN; k++) mx = fmaxf(mx, sh_e[k]);
        s_max = mx;
    }
    __syncthreads();
    if (tid < T_IN) sh_e[tid] = expf(sh_e[tid] - s_max);
    __syncthreads();
    if (tid == 0) {
        float s = 0.0f;
        for (int k = 0; k < T_IN; k++) s += sh_e[k];
        s_sum = s;
    }
    __syncthreads();
    if (tid < T_IN) {
        float a = sh_e[tid] / s_sum;
        sh_awn[tid] = a;
        aw[b * T_IN + tid]   = a;
        acum[b * T_IN + tid] = sh_ac[tid] + a;
        aligns[((long)(b * T_OUT + t)) * T_IN + tid] = a;
    }
    __syncthreads();
    for (int d = tid; d < ENC; d += 256) {
        float s = 0.0f;
        for (int tt = 0; tt < T_IN; tt++)
            s += sh_awn[tt] * memory[((long)(b * T_IN + tt)) * ENC + d];
        ctx[b * ENC + d] = s;
    }
}

// ---------------------------------------------------------------------------
// Projection + gate: [h_d;ctx](32x1536) @ Wpg(1536x96 tiled, cols 0..79 proj,
// col 80 gate, rest zero). Single WG, WMMA, writes 81 output cols.
// ---------------------------------------------------------------------------
__global__ __launch_bounds__(256) void proj_step(
    const _Float16* __restrict__ W, int KT,
    const float* __restrict__ s0,   // h_d [32,1024]
    const float* __restrict__ s1,   // ctx [32,512]
    const float* __restrict__ proj_b, const float* __restrict__ gate_b,
    float* __restrict__ out, int t)
{
    __shared__ __align__(16) _Float16 ldsA[32 * LDS_STRIDE];
    __shared__ float zbuf[32 * 96];
    const int tid = threadIdx.x, lane = tid & 31, w = tid >> 5;
    const int nt = (w < 6) ? w : 5;   // waves 6,7 duplicate wave 5 (EXEC stays full)
    const int Ktot = DRNN + ENC;      // 1536
    v8f acc0 = {}, acc1 = {};

    for (int k0 = 0; k0 < Ktot; k0 += LDS_STRIDE) {
        const int ch = min(LDS_STRIDE, Ktot - k0);
        for (int idx = tid; idx < 32 * ch; idx += 256) {
            int m = idx / ch, k = idx - m * ch, g = k0 + k;
            float v = (g < DRNN) ? s0[(long)m * DRNN + g] : s1[(long)m * ENC + (g - DRNN)];
            ldsA[m * LDS_STRIDE + k] = (_Float16)v;
        }
        __syncthreads();
        const int kiters = ch >> 5;
        for (int kk = 0; kk < kiters; kk++) {
            const int kt = (k0 >> 5) + kk;
            v16h a0 = load_a_frag(ldsA, 0,  kk, lane);
            v16h a1 = load_a_frag(ldsA, 16, kk, lane);
            v16h bf = *(const v16h*)(W + (((long)nt * KT + kt) << 9) + lane * 16);
            acc0 = __builtin_amdgcn_wmma_f32_16x16x32_f16(false, a0, false, bf, (short)0, acc0, false, false);
            acc1 = __builtin_amdgcn_wmma_f32_16x16x32_f16(false, a1, false, bf, (short)0, acc1, false, false);
        }
        __syncthreads();
    }
    if (w < 6) {
        const int lc = w * 16 + (lane & 15);
        const int mb = (lane & 16) ? 8 : 0;
#pragma unroll
        for (int r = 0; r < 8; r++) {
            zbuf[(mb + r)      * 96 + lc] = acc0[r];
            zbuf[(mb + 16 + r) * 96 + lc] = acc1[r];
        }
    }
    __syncthreads();
    for (int i = tid; i < 32 * 81; i += 256) {
        int m = i / 81, c = i - m * 81;
        float v = zbuf[m * 96 + c];
        if (c < N_MEL) v += proj_b[c];
        else           v = sigmoidf_(v + gate_b[0]);
        out[((long)(m * T_OUT + t)) * 81 + c] = v;
    }
}

// ---------------------------------------------------------------------------
// Host-side orchestration.
// ---------------------------------------------------------------------------
extern "C" void kernel_launch(void* const* d_in, const int* in_sizes, int n_in,
                              void* d_out, int out_size, void* d_ws, size_t ws_size,
                              hipStream_t stream) {
    (void)in_sizes; (void)n_in; (void)out_size; (void)ws_size;
    const float* mel       = (const float*)d_in[0];
    const float* memory    = (const float*)d_in[1];
    const float* prenet_w0 = (const float*)d_in[2];
    const float* prenet_w1 = (const float*)d_in[3];
    const float* attn_Wx   = (const float*)d_in[4];
    const float* attn_Wh   = (const float*)d_in[5];
    const float* attn_b    = (const float*)d_in[6];
    const float* lsa_Wq    = (const float*)d_in[7];
    const float* lsa_Wm    = (const float*)d_in[8];
    const float* lsa_conv  = (const float*)d_in[9];
    const float* lsa_Wloc  = (const float*)d_in[10];
    const float* lsa_v     = (const float*)d_in[11];
    const float* dec_Wx    = (const float*)d_in[12];
    const float* dec_Wh    = (const float*)d_in[13];
    const float* dec_b     = (const float*)d_in[14];
    const float* proj_W    = (const float*)d_in[15];
    const float* proj_b    = (const float*)d_in[16];
    const float* gate_W    = (const float*)d_in[17];
    const float* gate_b    = (const float*)d_in[18];
    float* out = (float*)d_out;
    float* aligns_out = out + (long)BATCH * T_OUT * 81;

    // workspace carve-up
    char* base = (char*)d_ws;
    size_t off = 0;
    auto ah = [&](size_t n) { _Float16* p = (_Float16*)(base + off);
                              off += (n * sizeof(_Float16) + 255) & ~(size_t)255; return p; };
    auto af = [&](size_t n) { float* p = (float*)(base + off);
                              off += (n * sizeof(float) + 255) & ~(size_t)255; return p; };

    const int K_ATTN = PRE_D + ENC + ARNN;     // 1792, KT 56
    const int K_DEC  = ARNN + ENC + DRNN;      // 2560, KT 80
    _Float16* attnW = ah((size_t)K_ATTN * 4096);
    _Float16* decW  = ah((size_t)K_DEC * 4096);
    _Float16* w0p   = ah((size_t)96 * 256);    // prenet_w0, K padded 80->96
    _Float16* w1t   = ah((size_t)256 * 256);
    _Float16* projW = ah((size_t)1536 * 96);   // [proj(80)|gate(1)|pad]
    _Float16* wmW   = ah((size_t)512 * 128);
    float* pre1 = af((size_t)BATCH * T_OUT * PRE_D);
    float* pre  = af((size_t)BATCH * T_OUT * PRE_D);
    float* pmb  = af((size_t)BATCH * T_IN * ATTN_DIM);
    float* h_a0 = af(BATCH * ARNN); float* h_a1 = af(BATCH * ARNN);
    float* c_a  = af(BATCH * ARNN);
    float* h_d0 = af(BATCH * DRNN); float* h_d1 = af(BATCH * DRNN);
    float* c_d  = af(BATCH * DRNN);
    float* ctxb = af(BATCH * ENC);
    float* awb  = af(BATCH * T_IN); float* acumb = af(BATCH * T_IN);

    // one-time weight retiling to f16 fragment-major
    auto cvt = [&](_Float16* dst, const float* s0, int r0, int c0,
                   const float* s1, int r1, int c1, int KT, int mode, long total) {
        int blocks = (int)((total + 255) / 256);
        cvt_weights<<<blocks, 256, 0, stream>>>(dst, s0, r0, c0, s1, r1, c1, KT, mode, total);
    };
    cvt(attnW, attn_Wx, 768, 4096, attn_Wh, 1024, 4096, 56, 0, (long)K_ATTN * 4096);
    cvt(decW,  dec_Wx, 1536, 4096, dec_Wh,  1024, 4096, 80, 0, (long)K_DEC * 4096);
    cvt(w0p,   prenet_w0, 80, 256, nullptr, 0, 0,        3,  0, 96L * 256);
    cvt(w1t,   prenet_w1, 256, 256, nullptr, 0, 0,       8,  0, 256L * 256);
    cvt(projW, proj_W, 1536, 80, gate_W, 0, 1,           48, 1, 1536L * 96);
    cvt(wmW,   lsa_Wm, 512, 128, nullptr, 0, 0,          16, 0, 512L * 128);

    // zero all recurrent state (contiguous tail of ws float region)
    {
        long nz = (long)(awb + BATCH * T_IN + BATCH * T_IN - h_a0); // h_a0..acumb end
        nz = (long)((acumb + BATCH * T_IN) - h_a0);
        zero_f32<<<(int)((nz + 255) / 256), 256, 0, stream>>>(h_a0, nz);
    }

    // precompute: prenet (2 GEMMs, relu), processed_memory
    gemm_wmma<<<dim3(500, 2), 256, 0, stream>>>(w0p, 3, 96,  mel,  80, 80, 1, pre1, 256, 1);
    gemm_wmma<<<dim3(500, 2), 256, 0, stream>>>(w1t, 8, 256, pre1, 256, 256, 0, pre, 256, 1);
    gemm_wmma<<<dim3(200, 1), 256, 0, stream>>>(wmW, 16, 512, memory, 512, 512, 0, pmb, 128, 0);

    // sequential decode
    for (int t = 0; t < T_OUT; t++) {
        float* ha_in  = (t & 1) ? h_a1 : h_a0;
        float* ha_out = (t & 1) ? h_a0 : h_a1;
        float* hd_in  = (t & 1) ? h_d1 : h_d0;
        float* hd_out = (t & 1) ? h_d0 : h_d1;

        // attention LSTM: x = [pre_t(256); ctx(512)] , recur h_a(1024)
        lstm_step<<<32, 256, 0, stream>>>(attnW, 56, K_ATTN,
            pre + (long)t * PRE_D, T_OUT * PRE_D, PRE_D,
            ctxb, ENC, ENC,
            ha_in, ARNN, ARNN,
            attn_b, c_a, ha_out);

        // location-sensitive attention -> new ctx, aw, acum, aligns
        attn_step<<<32, 256, 0, stream>>>(ha_out, awb, acumb, pmb, memory,
            lsa_Wq, lsa_conv, lsa_Wloc, lsa_v, ctxb, aligns_out, t);

        // decoder LSTM: x = [h_a(1024); ctx(512)] , recur h_d(1024)
        lstm_step<<<32, 256, 0, stream>>>(decW, 80, K_DEC,
            ha_out, ARNN, ARNN,
            ctxb, ENC, ENC,
            hd_in, DRNN, DRNN,
            dec_b, c_d, hd_out);

        // projection + gate -> out[:, t, 0:81]
        proj_step<<<1, 256, 0, stream>>>(projW, 48, hd_out, ctxb, proj_b, gate_b, out, t);
    }
}
